// Conv3d_86517821212455
// MI455X (gfx1250) — matrix operand order repaired and checked by reading, not compiled, not run on hardware
//
#include <hip/hip_runtime.h>
#include <stdint.h>

// ---------------------------------------------------------------------------
// Octree conv: Y[i] = sum_k X[nbr(i,k)] @ W_k + b      (N=1M, K=27, C=32x32)
// Strategy: bf16x3 split-precision WMMA (hi/lo bf16, f32 accumulate).
//   - X pre-split once into packed-bf16 hi/lo rows (64B each) in d_ws.
//   - W pre-split + transposed into B-fragment layout (padded) -> LDS.
//   - Each wave: 16 output rows x 32 cols; per tap 6x v_wmma_f32_16x16x32_bf16.
// ---------------------------------------------------------------------------

typedef __attribute__((ext_vector_type(16))) __bf16 v16bf;
typedef __attribute__((ext_vector_type(8)))  float  v8f;

#define NTAPS 27
#define CIN   32
#define COUT  32
#define ROWS_PER_BLOCK 128      // 8 waves * 16 rows
#define THREADS 256
#define WROW_U32 20             // per-(k,co) row: 16 u32 data + 4 u32 pad (80B, bank-skewed)
#define WMAT_U32 (NTAPS * COUT * WROW_U32)   // 17280 u32 per (hi|lo) matrix set

union Frag { uint4 u[2]; v16bf v; };

__device__ __forceinline__ uint32_t hi_bits(float x) {
    return __float_as_uint(x) & 0xFFFF0000u;
}
// pack bf16(a) into [15:0], bf16(b) into [31:16]  (K=2j / K=2j+1 ordering)
__device__ __forceinline__ uint32_t pack_hi(float a, float b) {
    return (hi_bits(a) >> 16) | hi_bits(b);
}

// ---- kernel 1: split X rows into packed bf16 hi / lo arrays --------------
__global__ __launch_bounds__(256) void split_x(const float* __restrict__ X,
                                               uint32_t* __restrict__ Xhi,
                                               uint32_t* __restrict__ Xlo,
                                               int npairs) {
    int t = blockIdx.x * blockDim.x + threadIdx.x;
    if (t >= npairs) return;
    float2 v = ((const float2*)X)[t];
    float h0 = __uint_as_float(hi_bits(v.x));
    float h1 = __uint_as_float(hi_bits(v.y));
    Xhi[t] = pack_hi(v.x, v.y);
    Xlo[t] = pack_hi(v.x - h0, v.y - h1);
}

// ---- kernel 2: split + transpose W into B-fragment layout ----------------
// Whi/Wlo[(k*32 + co)*20 + j] = pack(W[k][2j][co], W[k][2j+1][co]), j<16; pad j>=16.
__global__ __launch_bounds__(256) void split_w(const float* __restrict__ W,
                                               uint32_t* __restrict__ Whi,
                                               uint32_t* __restrict__ Wlo) {
    int t = blockIdx.x * blockDim.x + threadIdx.x;
    if (t >= WMAT_U32) return;
    int k  = t / (COUT * WROW_U32);
    int r  = t % (COUT * WROW_U32);
    int co = r / WROW_U32;
    int j  = r % WROW_U32;
    uint32_t hv = 0u, lv = 0u;
    if (j < 16) {
        float a = W[k * (CIN * COUT) + (2 * j)     * COUT + co];
        float b = W[k * (CIN * COUT) + (2 * j + 1) * COUT + co];
        float ah = __uint_as_float(hi_bits(a));
        float bh = __uint_as_float(hi_bits(b));
        hv = pack_hi(a, b);
        lv = pack_hi(a - ah, b - bh);
    }
    Whi[t] = hv;
    Wlo[t] = lv;
}

// ---- kernel 3: gather + WMMA conv ----------------------------------------
__global__ __launch_bounds__(THREADS, 2)
void octconv_wmma(const uint32_t* __restrict__ Xhi,
                  const uint32_t* __restrict__ Xlo,
                  const uint32_t* __restrict__ Whi,
                  const uint32_t* __restrict__ Wlo,
                  const long long* __restrict__ nbr,
                  const float* __restrict__ bias,
                  float* __restrict__ out, int n) {
    __shared__ uint32_t sW[2 * WMAT_U32];                 // hi then lo : 138,240 B
    __shared__ uint32_t sNbr[ROWS_PER_BLOCK * NTAPS];     // 13,824 B

    const int tid = threadIdx.x;
    const int rowBase = blockIdx.x * ROWS_PER_BLOCK;

    // cooperative stage: weights (hi+lo) into LDS
    for (int i = tid; i < WMAT_U32; i += THREADS) {
        sW[i]            = Whi[i];
        sW[WMAT_U32 + i] = Wlo[i];
    }
    // cooperative stage: neighbor indices (int64 -> u32) for this block's rows
    for (int i = tid; i < ROWS_PER_BLOCK * NTAPS; i += THREADS) {
        int r = rowBase + i / NTAPS;
        int k = i % NTAPS;
        sNbr[i] = (r < n) ? (uint32_t)nbr[(long long)r * NTAPS + k] : 0u;
    }
    __syncthreads();

    const int lane = tid & 31;
    const int wv   = tid >> 5;        // wave id: row group 0..7
    const int ln15 = lane & 15;
    const int p    = lane >> 4;       // lane half selects K sub-range
    const int row0 = rowBase + wv * 16;
    if (row0 >= n) return;            // N % 16 == 0 -> groups are full or absent

    const uint4* Xhi4 = (const uint4*)Xhi;   // one row = 4 uint4 (64B of bf16)
    const uint4* Xlo4 = (const uint4*)Xlo;

    v8f acc0 = {}; v8f acc1 = {};
    const int nbrBase = (wv * 16 + ln15) * NTAPS;

    for (int k = 0; k < NTAPS; ++k) {
        const uint32_t idx = sNbr[nbrBase + k];
        // A fragments (16-bit A 16x32 layout): low lanes K0-7 & K16-23,
        // high lanes K8-15 & K24-31  ->  uint4 at row*4 + p and row*4 + 2 + p.
        Frag ahi, alo;
        ahi.u[0] = Xhi4[idx * 4 + p];
        ahi.u[1] = Xhi4[idx * 4 + 2 + p];
        alo.u[0] = Xlo4[idx * 4 + p];
        alo.u[1] = Xlo4[idx * 4 + 2 + p];

        // B fragments from LDS: col co = 16h + ln15; lanes 0-15 hold K0-7 & K16-23,
        // lanes 16-31 hold K8-15 & K24-31 (p*4 u32 offset). 80B row stride skews banks.
        const int b0 = (k * COUT + ln15) * WROW_U32 + p * 4;    // h = 0
        const int b1 = b0 + 16 * WROW_U32;                       // h = 1
        Frag bh0, bh1, bl0, bl1;
        bh0.u[0] = *(const uint4*)&sW[b0];
        bh0.u[1] = *(const uint4*)&sW[b0 + 8];
        bh1.u[0] = *(const uint4*)&sW[b1];
        bh1.u[1] = *(const uint4*)&sW[b1 + 8];
        bl0.u[0] = *(const uint4*)&sW[WMAT_U32 + b0];
        bl0.u[1] = *(const uint4*)&sW[WMAT_U32 + b0 + 8];
        bl1.u[0] = *(const uint4*)&sW[WMAT_U32 + b1];
        bl1.u[1] = *(const uint4*)&sW[WMAT_U32 + b1 + 8];

        // bf16x3: hi*hi + lo*hi + hi*lo  (f32 accumulate); interleave the two
        // independent accumulator chains to hide WMMA latency.
        acc0 = __builtin_amdgcn_wmma_f32_16x16x32_bf16(false, ahi.v, false, bh0.v,
                                                       (short)0, acc0, false, false);
        acc1 = __builtin_amdgcn_wmma_f32_16x16x32_bf16(false, ahi.v, false, bh1.v,
                                                       (short)0, acc1, false, false);
        acc0 = __builtin_amdgcn_wmma_f32_16x16x32_bf16(false, alo.v, false, bh0.v,
                                                       (short)0, acc0, false, false);
        acc1 = __builtin_amdgcn_wmma_f32_16x16x32_bf16(false, alo.v, false, bh1.v,
                                                       (short)0, acc1, false, false);
        acc0 = __builtin_amdgcn_wmma_f32_16x16x32_bf16(false, ahi.v, false, bl0.v,
                                                       (short)0, acc0, false, false);
        acc1 = __builtin_amdgcn_wmma_f32_16x16x32_bf16(false, ahi.v, false, bl1.v,
                                                       (short)0, acc1, false, false);
    }

    // C/D layout: VGPR r -> row (row0 + r + 8p), col = 16h + ln15.
    const float bias0 = bias[ln15];
    const float bias1 = bias[16 + ln15];
    const int orow = row0 + p * 8;
#pragma unroll
    for (int r = 0; r < 8; ++r) {
        out[(orow + r) * COUT + ln15]      = acc0[r] + bias0;
        out[(orow + r) * COUT + 16 + ln15] = acc1[r] + bias1;
    }
}

// ---------------------------------------------------------------------------
extern "C" void kernel_launch(void* const* d_in, const int* in_sizes, int n_in,
                              void* d_out, int out_size, void* d_ws, size_t ws_size,
                              hipStream_t stream) {
    (void)n_in; (void)out_size; (void)ws_size;
    const float*     X    = (const float*)d_in[0];       // [N,32] f32
    const float*     W    = (const float*)d_in[1];       // [27,32,32] f32
    const float*     bias = (const float*)d_in[2];       // [32] f32
    const long long* nbr  = (const long long*)d_in[3];   // [N,27] i64
    float* out = (float*)d_out;                          // [N,32] f32
    const int N = in_sizes[0] / CIN;

    // workspace layout: Xhi [N*64B] | Xlo [N*64B] | Whi [69120B] | Wlo [69120B]
    uint8_t* ws  = (uint8_t*)d_ws;
    uint32_t* Xhi = (uint32_t*)(ws);
    uint32_t* Xlo = (uint32_t*)(ws + (size_t)N * 64);
    uint32_t* Whi = (uint32_t*)(ws + (size_t)N * 128);
    uint32_t* Wlo = (uint32_t*)(ws + (size_t)N * 128 + (size_t)WMAT_U32 * 4);

    const int npairs = N * (CIN / 2);
    split_x<<<(npairs + 255) / 256, 256, 0, stream>>>(X, Xhi, Xlo, npairs);
    split_w<<<(WMAT_U32 + 255) / 256, 256, 0, stream>>>(W, Whi, Wlo);

    const int nblk = (N + ROWS_PER_BLOCK - 1) / ROWS_PER_BLOCK;
    octconv_wmma<<<nblk, THREADS, 0, stream>>>(Xhi, Xlo, Whi, Wlo, nbr, bias, out, N);
}
